// MHAEinsum_3143916060977
// MI455X (gfx1250) — compile-verified, matrix operand order checked by reading, not run on hardware
//
#include <hip/hip_runtime.h>
#include <math.h>
#include <stdint.h>

// ---------------------------------------------------------------------------
// MHA forward for MI455X (gfx1250): bf16 WMMA everywhere, flash-style fused
// attention, Tensor Data Mover (TDM) for strided-tile -> LDS staging.
// ---------------------------------------------------------------------------

#define B_  4
#define N_  2048
#define D_  1024
#define H_  16
#define HD_ 64

typedef __attribute__((ext_vector_type(16))) __bf16    v16bf;
typedef __attribute__((ext_vector_type(8)))  float     v8f;
typedef __attribute__((ext_vector_type(4)))  uint32_t  u32x4;
typedef __attribute__((ext_vector_type(4)))  int       i32x4;
typedef __attribute__((ext_vector_type(8)))  int       i32x8;

#if __has_builtin(__builtin_amdgcn_tensor_load_to_lds)
#define HAVE_TDM 1
#else
#define HAVE_TDM 0
#endif

#if HAVE_TDM
// 2D tile DMA: global (d1 rows x d0 contiguous elems, row stride in elems,
// 2-byte elements) -> LDS packed row-major. Descriptor per CDNA5 ISA §8.
// Issue from ONE wave only; EXEC is ignored by TDM; completion on TENSORcnt.
__device__ __forceinline__ void tdm_load_2d(const void* gptr, uint32_t lds_off,
                                            uint32_t d0, uint32_t d1,
                                            uint32_t stride_elems) {
    uint64_t ga = (uint64_t)(uintptr_t)gptr;
    u32x4 g0;
    g0[0] = 1u;                                          // count=1 (valid), user
    g0[1] = lds_off;                                     // lds_addr (bytes)
    g0[2] = (uint32_t)ga;                                // global_addr[31:0]
    g0[3] = (uint32_t)((ga >> 32) & 0x01FFFFFFu)         // global_addr[56:32]
          | (2u << 30);                                  // type = 2 ("image")
    i32x8 g1;
    g1[0] = (int)(1u << 16);                             // data_size = 2 bytes
    g1[1] = (int)((d0 & 0xFFFFu) << 16);                 // tensor_dim0[15:0]
    g1[2] = (int)(((d0 >> 16) & 0xFFFFu)                 // tensor_dim0[31:16]
          | ((d1 & 0xFFFFu) << 16));                     // tensor_dim1[15:0]
    g1[3] = (int)(((d1 >> 16) & 0xFFFFu)                 // tensor_dim1[31:16]
          | ((d0 & 0xFFFFu) << 16));                     // tile_dim0
    g1[4] = (int)(d1 & 0xFFFFu);                         // tile_dim1 (tile_dim2=0)
    g1[5] = (int)stride_elems;                           // tensor_dim0_stride lo
    g1[6] = 0;                                           // stride hi | dim1_stride lo
    g1[7] = 0;
    i32x4 z4 = {0, 0, 0, 0};
#if defined(__clang_major__) && __clang_major__ >= 23
    i32x8 z8 = {0, 0, 0, 0, 0, 0, 0, 0};
    __builtin_amdgcn_tensor_load_to_lds(g0, g1, z4, z4, z8, 0);
#else
    __builtin_amdgcn_tensor_load_to_lds(g0, g1, z4, z4, 0);
#endif
}
#endif

// ---------------------------------------------------------------------------
// fp32 -> bf16 conversion
// ---------------------------------------------------------------------------
__global__ __launch_bounds__(256) void f2bf_kernel(const float* __restrict__ in,
                                                   __bf16* __restrict__ out, int n) {
    int i = blockIdx.x * 256 + threadIdx.x;
    if (i < n) out[i] = (__bf16)in[i];
}

// ---------------------------------------------------------------------------
// Tiled bf16 GEMM:  C[M x Nc] = alpha * A[M x K] * B  (+ bias)
//   transB == 0 : B is [K  x Nc] row-major  (C = A*B)
//   transB == 1 : B is [Nc x K ] row-major  (C = A*B^T)
// BM=BN=128, BK=32. 256 threads = 8 waves in a 2x4 (M x N) wave grid.
// A tile (and B tile when transB) staged by TDM; transB==0 B tile needs a
// transpose so it stays on the manual VMEM+DS-scatter path.
// ---------------------------------------------------------------------------
template <int OUTBF>
__global__ __launch_bounds__(256) void gemm_bf16_wmma(
    const __bf16* __restrict__ A, const __bf16* __restrict__ Bm,
    void* __restrict__ Cout, const float* __restrict__ bias,
    float alpha, int M, int Nc, int K, int transB)
{
    __shared__ __align__(32) __bf16 sA [128][32];
    __shared__ __align__(32) __bf16 sBT[128][32];   // stored transposed: [n][k]

    const int m0 = blockIdx.y * 128;
    const int n0 = blockIdx.x * 128;
    const int t    = threadIdx.x;
    const int wave = t >> 5, lane = t & 31;
    const int lo   = lane & 15, hi = lane >> 4;
    const int wm   = wave >> 2, wn = wave & 3;

    v8f acc[4][2] = {};

    for (int k0 = 0; k0 < K; k0 += 32) {
        __syncthreads();
#if HAVE_TDM
        if (t < 32) {                         // wave 0 drives the TDM
            tdm_load_2d(&A[(size_t)m0 * K + k0],
                        (uint32_t)(uintptr_t)&sA[0][0], 32, 128, (uint32_t)K);
            if (transB)
                tdm_load_2d(&Bm[(size_t)n0 * K + k0],
                            (uint32_t)(uintptr_t)&sBT[0][0], 32, 128, (uint32_t)K);
        }
#else
        {
            int r = t >> 1, half = (t & 1) * 16;
            *(v16bf*)&sA[r][half] =
                *(const v16bf*)&A[(size_t)(m0 + r) * K + k0 + half];
            __builtin_prefetch(&A[(size_t)(m0 + r) * K + k0 + 32 + half], 0, 3);
        }
        if (transB) {
            int c = t >> 1, half = (t & 1) * 16;
            *(v16bf*)&sBT[c][half] =
                *(const v16bf*)&Bm[(size_t)(n0 + c) * K + k0 + half];
        }
#endif
        if (!transB) {                        // transpose scatter (no TDM transpose)
            int kr = t >> 3, cb = (t & 7) * 16;
            v16bf v = *(const v16bf*)&Bm[(size_t)(k0 + kr) * Nc + n0 + cb];
            __builtin_prefetch(&Bm[(size_t)(k0 + 32 + kr) * Nc + n0 + cb], 0, 3);
            #pragma unroll
            for (int i = 0; i < 16; ++i) sBT[cb + i][kr] = v[i];
        }
#if HAVE_TDM
        if (t < 32) __builtin_amdgcn_s_wait_tensorcnt(0);
#endif
        __syncthreads();

        // ---- fragments first, then 8 back-to-back WMMAs ----
        v16bf af[4], bfr[2];
        #pragma unroll
        for (int mt = 0; mt < 4; ++mt)
            af[mt] = *(const v16bf*)&sA[wm * 64 + mt * 16 + lo][hi * 16];
        #pragma unroll
        for (int nt = 0; nt < 2; ++nt)
            bfr[nt] = *(const v16bf*)&sBT[wn * 32 + nt * 16 + lo][hi * 16];
        #pragma unroll
        for (int mt = 0; mt < 4; ++mt)
            #pragma unroll
            for (int nt = 0; nt < 2; ++nt)
                acc[mt][nt] = __builtin_amdgcn_wmma_f32_16x16x32_bf16(
                    false, af[mt], false, bfr[nt], (short)0, acc[mt][nt],
                    false, false);
    }

    // ---- epilogue: C layout = VGPR r, lanes 0-15 -> row r, lanes 16-31 -> r+8
    #pragma unroll
    for (int mt = 0; mt < 4; ++mt)
        #pragma unroll
        for (int nt = 0; nt < 2; ++nt)
            #pragma unroll
            for (int r = 0; r < 8; ++r) {
                int grow = m0 + wm * 64 + mt * 16 + r + hi * 8;
                int gcol = n0 + wn * 32 + nt * 16 + lo;
                float v = acc[mt][nt][r] * alpha;
                if (bias) v += bias[gcol];
                if (OUTBF)
                    ((__bf16*)Cout)[(size_t)grow * Nc + gcol] = (__bf16)v;
                else
                    ((float*)Cout)[(size_t)grow * Nc + gcol] = v;
            }
}

// ---------------------------------------------------------------------------
// butterfly reductions over the 16-lane half-wave that owns one C-matrix row
// (wave32: masks 1..8 never cross the 16-lane groups)
// ---------------------------------------------------------------------------
__device__ __forceinline__ float rowmax16(float v) {
    #pragma unroll
    for (int m = 8; m >= 1; m >>= 1) v = fmaxf(v, __shfl_xor(v, m, 32));
    return v;
}
__device__ __forceinline__ float rowsum16(float v) {
    #pragma unroll
    for (int m = 8; m >= 1; m >>= 1) v += __shfl_xor(v, m, 32);
    return v;
}

// ---------------------------------------------------------------------------
// Fused causal flash attention.
//   Q,K,V layout: [B*N, D] bf16 with column = h*HD + hd. Q pre-scaled 1/sqrt(HD).
// Grid: x = 128-row query tile, y = b*H + h. 256 thr = 8 waves, 16 rows/wave.
// K tile staged by TDM; V tile transposed manually; P repacked via per-wave
// LDS scratch guarded by s_wait_dscnt.
// ---------------------------------------------------------------------------
__global__ __launch_bounds__(256) void flash_attn_kernel(
    const __bf16* __restrict__ Q, const __bf16* __restrict__ Kt,
    const __bf16* __restrict__ V, __bf16* __restrict__ Ctx)
{
    __shared__ __align__(32) __bf16 ldsK [32][64];    // [key][hd]
    __shared__ __align__(32) __bf16 ldsVT[64][32];    // [hd][key]
    __shared__ __align__(32) __bf16 ldsP [8][16][32]; // per-wave P scratch

    const int bh = blockIdx.y;
    const int b  = bh >> 4, h = bh & 15;
    const size_t base = (size_t)b * N_ * D_ + (size_t)h * HD_;
    const int q0 = blockIdx.x * 128;

    const int t    = threadIdx.x;
    const int wave = t >> 5, lane = t & 31;
    const int lo   = lane & 15, hi = lane >> 4;
    const int qrb  = q0 + wave * 16;

    // Q fragments: 16 rows x 64 hd = two 16x32 A-fragments, loaded once.
    v16bf qf[2];
    #pragma unroll
    for (int c = 0; c < 2; ++c)
        qf[c] = *(const v16bf*)&Q[base + (size_t)(qrb + lo) * D_ + c * 32 + hi * 16];

    v8f   acc[4] = {};
    float m_run[8], l_run[8];
    #pragma unroll
    for (int r = 0; r < 8; ++r) { m_run[r] = -INFINITY; l_run[r] = 0.f; }

    const int ktiles = (q0 + 128) >> 5;       // causal bound for this block

    for (int kt = 0; kt < ktiles; ++kt) {
        const int kb = kt * 32;
        __syncthreads();                       // protect LDS reuse
#if HAVE_TDM
        if (t < 32)                            // wave 0 drives the K-tile DMA
            tdm_load_2d(&Kt[base + (size_t)kb * D_],
                        (uint32_t)(uintptr_t)&ldsK[0][0], 64, 32, D_);
#else
        if (t < 128) {
            int r = t >> 2, cb = (t & 3) * 16;
            *(v16bf*)&ldsK[r][cb] =
                *(const v16bf*)&Kt[base + (size_t)(kb + r) * D_ + cb];
        }
#endif
        if (t >= 128) {                        // V tile: transpose into [hd][key]
            int t2 = t - 128;
            int kr = t2 >> 2, cb = (t2 & 3) * 16;
            v16bf v = *(const v16bf*)&V[base + (size_t)(kb + kr) * D_ + cb];
            #pragma unroll
            for (int i = 0; i < 16; ++i) ldsVT[cb + i][kr] = v[i];
        }
#if HAVE_TDM
        if (t < 32) __builtin_amdgcn_s_wait_tensorcnt(0);
#endif
        __syncthreads();

        // ---- S = Q * K^T : preload all 4 B-fragments, then 4 WMMAs ----
        v16bf kf[4];
        kf[0] = *(const v16bf*)&ldsK[lo     ][     hi * 16];
        kf[1] = *(const v16bf*)&ldsK[16 + lo][     hi * 16];
        kf[2] = *(const v16bf*)&ldsK[lo     ][32 + hi * 16];
        kf[3] = *(const v16bf*)&ldsK[16 + lo][32 + hi * 16];
        v8f s0 = {}, s1 = {};
        s0 = __builtin_amdgcn_wmma_f32_16x16x32_bf16(false, qf[0], false, kf[0],
                                                     (short)0, s0, false, false);
        s1 = __builtin_amdgcn_wmma_f32_16x16x32_bf16(false, qf[0], false, kf[1],
                                                     (short)0, s1, false, false);
        s0 = __builtin_amdgcn_wmma_f32_16x16x32_bf16(false, qf[1], false, kf[2],
                                                     (short)0, s0, false, false);
        s1 = __builtin_amdgcn_wmma_f32_16x16x32_bf16(false, qf[1], false, kf[3],
                                                     (short)0, s1, false, false);

        // ---- causal mask + online softmax ----
        #pragma unroll
        for (int r = 0; r < 8; ++r) {
            int row  = qrb + r + hi * 8;
            int key0 = kb + lo, key1 = kb + 16 + lo;
            float a = (key0 > row) ? -INFINITY : s0[r];
            float c = (key1 > row) ? -INFINITY : s1[r];
            float rm   = rowmax16(fmaxf(a, c));
            float newm = fmaxf(m_run[r], rm);
            float f    = __expf(m_run[r] - newm);   // 0 on first tile
            m_run[r]   = newm;
            float p0 = __expf(a - newm);
            float p1 = __expf(c - newm);
            l_run[r] = l_run[r] * f + rowsum16(p0 + p1);
            #pragma unroll
            for (int ct = 0; ct < 4; ++ct) acc[ct][r] *= f;
            ldsP[wave][r + hi * 8][lo]      = (__bf16)p0;
            ldsP[wave][r + hi * 8][16 + lo] = (__bf16)p1;
        }
        // in-wave LDS RAW: drain the DS counter before reading P back
        asm volatile("s_wait_dscnt 0x0" ::: "memory");

        // ---- ctx += P * V : 4 WMMAs (n over hd, k = 32 keys) ----
        v16bf pf = *(const v16bf*)&ldsP[wave][lo][hi * 16];
        #pragma unroll
        for (int ct = 0; ct < 4; ++ct) {
            v16bf vf = *(const v16bf*)&ldsVT[ct * 16 + lo][hi * 16];
            acc[ct] = __builtin_amdgcn_wmma_f32_16x16x32_bf16(
                false, pf, false, vf, (short)0, acc[ct], false, false);
        }
    }

    // ---- normalize and write ctx (bf16, merged-head layout [B*N, D]) ----
    #pragma unroll
    for (int ct = 0; ct < 4; ++ct)
        #pragma unroll
        for (int r = 0; r < 8; ++r) {
            int   row = qrb + r + hi * 8;
            float v   = acc[ct][r] / l_run[r];
            Ctx[base + (size_t)row * D_ + ct * 16 + lo] = (__bf16)v;
        }
}

// ---------------------------------------------------------------------------
// launch
// ---------------------------------------------------------------------------
extern "C" void kernel_launch(void* const* d_in, const int* in_sizes, int n_in,
                              void* d_out, int out_size, void* d_ws, size_t ws_size,
                              hipStream_t stream) {
    const float* x  = (const float*)d_in[0];
    const float* Wq = (const float*)d_in[1];
    const float* Wk = (const float*)d_in[2];
    const float* Wv = (const float*)d_in[3];
    const float* Wo = (const float*)d_in[4];
    const float* bo = (const float*)d_in[5];
    float* out = (float*)d_out;

    const size_t nTok = (size_t)B_ * N_;          // 8192
    const size_t nXD  = nTok * D_;                // 8 Mi elements
    const size_t nWW  = (size_t)D_ * D_;          // 1 Mi elements

    char*  ws  = (char*)d_ws;
    size_t off = 0;
    auto carve = [&](size_t elems) {
        char* p = ws + off;
        off = (off + elems * sizeof(__bf16) + 255) & ~(size_t)255;
        return (__bf16*)p;
    };
    __bf16* xb   = carve(nXD);
    __bf16* wqb  = carve(nWW);
    __bf16* wkb  = carve(nWW);
    __bf16* wvb  = carve(nWW);
    __bf16* wob  = carve(nWW);
    __bf16* Qb   = carve(nXD);
    __bf16* Kb   = carve(nXD);
    __bf16* Vb   = carve(nXD);
    __bf16* ctxb = carve(nXD);

    // 1) fp32 -> bf16
    f2bf_kernel<<<dim3((unsigned)((nXD + 255) / 256)), 256, 0, stream>>>(x,  xb,  (int)nXD);
    f2bf_kernel<<<dim3((unsigned)((nWW + 255) / 256)), 256, 0, stream>>>(Wq, wqb, (int)nWW);
    f2bf_kernel<<<dim3((unsigned)((nWW + 255) / 256)), 256, 0, stream>>>(Wk, wkb, (int)nWW);
    f2bf_kernel<<<dim3((unsigned)((nWW + 255) / 256)), 256, 0, stream>>>(Wv, wvb, (int)nWW);
    f2bf_kernel<<<dim3((unsigned)((nWW + 255) / 256)), 256, 0, stream>>>(Wo, wob, (int)nWW);

    // 2) QKV projections: [8192 x 1024] = xb * W   (Q scaled by 1/sqrt(HD))
    dim3 ggrid(D_ / 128, (unsigned)(nTok / 128));      // (8, 64)
    gemm_bf16_wmma<1><<<ggrid, 256, 0, stream>>>(xb, wqb, Qb, nullptr,
                                                 0.125f, (int)nTok, D_, D_, 0);
    gemm_bf16_wmma<1><<<ggrid, 256, 0, stream>>>(xb, wkb, Kb, nullptr,
                                                 1.0f, (int)nTok, D_, D_, 0);
    gemm_bf16_wmma<1><<<ggrid, 256, 0, stream>>>(xb, wvb, Vb, nullptr,
                                                 1.0f, (int)nTok, D_, D_, 0);

    // 3) fused causal flash attention -> ctx (bf16, merged heads)
    dim3 agrid(N_ / 128, B_ * H_);                     // (16, 64)
    flash_attn_kernel<<<agrid, 256, 0, stream>>>(Qb, Kb, Vb, ctxb);

    // 4) output projection: out = ctx @ Wo^T + bo  (fp32 out)
    gemm_bf16_wmma<0><<<ggrid, 256, 0, stream>>>(ctxb, wob, out, bo,
                                                 1.0f, (int)nTok, D_, D_, 1);
}
